// HybridConvLayer_50560355008634
// MI455X (gfx1250) — matrix operand order repaired and checked by reading, not compiled, not run on hardware
//
#include <hip/hip_runtime.h>
#include <math.h>

typedef __attribute__((ext_vector_type(16))) _Float16 v16h;
typedef __attribute__((ext_vector_type(8)))  _Float16 v8h;
typedef __attribute__((ext_vector_type(8)))  float    v8f;

#define F_DIM 128
#define O_DIM 128

// ---------------- setup kernels ----------------
__global__ void k_zero(float* p, int n) {
    int i = blockIdx.x * blockDim.x + threadIdx.x;
    if (i < n) p[i] = 0.f;
}

__global__ void k_deg(const long long* __restrict__ dst, float* __restrict__ deg, int E) {
    int e = blockIdx.x * blockDim.x + threadIdx.x;
    if (e < E) unsafeAtomicAdd(&deg[(int)dst[e]], 1.0f);
}

__global__ void k_dinv(const float* __restrict__ deg, float* __restrict__ dinv, int N) {
    int i = blockIdx.x * blockDim.x + threadIdx.x;
    if (i < N) {
        float d = deg[i];
        dinv[i] = (d > 0.f) ? rsqrtf(fmaxf(d, 1e-12f)) : 0.f;
    }
}

__global__ void k_edgew(const long long* __restrict__ src, const long long* __restrict__ dst,
                        const float* __restrict__ dinv, float* __restrict__ w, int E) {
    int e = blockIdx.x * blockDim.x + threadIdx.x;
    if (e < E) w[e] = dinv[(int)src[e]] * dinv[(int)dst[e]] * 0.5f;
}

// out = 0.5 * a  (self-loop term init), vectorized float4
__global__ void k_half(const float* __restrict__ a, float* __restrict__ out, int n4) {
    int i = blockIdx.x * blockDim.x + threadIdx.x;
    if (i < n4) {
        float4 v = ((const float4*)a)[i];
        v.x *= 0.5f; v.y *= 0.5f; v.z *= 0.5f; v.w *= 0.5f;
        ((float4*)out)[i] = v;
    }
}

// one wave per edge; lane handles 4 contiguous features.
// edge index forced wave-uniform -> scalar (SMEM) loads of src/dst/w.
__global__ void k_scatter(const long long* __restrict__ src, const long long* __restrict__ dst,
                          const float* __restrict__ w, const float* __restrict__ a,
                          float* __restrict__ out, int E) {
    int e = __builtin_amdgcn_readfirstlane(blockIdx.x * 8 + (threadIdx.x >> 5));
    int lane = threadIdx.x & 31;
    if (e >= E) return;                 // wave-uniform
    int s = (int)src[e];
    int d = (int)dst[e];
    float we = w[e];
    const float4 v = *(const float4*)(a + (size_t)s * F_DIM + lane * 4);
    float* op = out + (size_t)d * F_DIM + lane * 4;
    unsafeAtomicAdd(op + 0, we * v.x);
    unsafeAtomicAdd(op + 1, we * v.y);
    unsafeAtomicAdd(op + 2, we * v.z);
    unsafeAtomicAdd(op + 3, we * v.w);
}

// ---------------- weight packing into WMMA B-fragment layout ----------------
// layout: [c][ntile(8)][kstep(4)][lane(32)][h(16)] ; B element (K,Ncol):
// lane = (Ncol%16) + ((K%32)&8 ? 16 : 0), h = (K%8) + ((K%32)>=16 ? 8 : 0)
__global__ void k_packW(const float* __restrict__ Wlow, const float* __restrict__ Wband,
                        _Float16* __restrict__ wp, int total) {
    int t = blockIdx.x * blockDim.x + threadIdx.x;
    if (t >= total) return;
    int c = t / (F_DIM * O_DIM);
    int r = t % (F_DIM * O_DIM);
    int k = r / O_DIM;
    int n = r % O_DIM;
    float v = (c < 4) ? Wlow[c * F_DIM * O_DIM + r] : Wband[(c - 4) * F_DIM * O_DIM + r];
    int nt = n >> 4, ks = k >> 5, kr = k & 31;
    int lane = (n & 15) + ((kr & 8) ? 16 : 0);
    int h = (kr & 7) + ((kr & 16) ? 8 : 0);
    wp[((((size_t)c * 8 + nt) * 4 + ks) * 32 + lane) * 16 + h] = (_Float16)v;
}

// W_mlp: [256 x 128] -> [ntile(8)][kstep(8)][lane(32)][h(16)]
__global__ void k_packWmlp(const float* __restrict__ Wmlp, _Float16* __restrict__ wp, int total) {
    int t = blockIdx.x * blockDim.x + threadIdx.x;
    if (t >= total) return;
    int k = t / O_DIM;
    int n = t % O_DIM;
    int nt = n >> 4, ks = k >> 5, kr = k & 31;
    int lane = (n & 15) + ((kr & 8) ? 16 : 0);
    int h = (kr & 7) + ((kr & 16) ? 8 : 0);
    wp[(((size_t)nt * 8 + ks) * 32 + lane) * 16 + h] = (_Float16)Wmlp[t];
}

// ---------------- 7-channel feature GEMM via WMMA ----------------
// h[c] = feats[c] @ W[c] + b[c].  One wave owns a 16-row M tile and loops over
// all 8 N tiles, holding the A fragments (4 ksteps) in registers for reuse.
__global__ void k_gemm_feats(const float* __restrict__ x,  const float* __restrict__ a1,
                             const float* __restrict__ a2, const float* __restrict__ a4,
                             const _Float16* __restrict__ wp,
                             const float* __restrict__ blow, const float* __restrict__ bband,
                             _Float16* __restrict__ hH, int N) {
    int c    = blockIdx.y;
    int wave = threadIdx.x >> 5;
    int lane = threadIdx.x & 31;
    int m0 = (blockIdx.x * 4 + wave) * 16;
    if (m0 >= N) return;   // wave-uniform

    const float* p1; const float* p2 = nullptr;
    switch (c) {
        case 0: p1 = x;  break;
        case 1: p1 = a1; break;
        case 2: p1 = a2; break;
        case 3: p1 = a4; break;
        case 4: p1 = x;  p2 = a1; break;
        case 5: p1 = a1; p2 = a2; break;
        default: p1 = a2; p2 = a4; break;
    }

    int row  = m0 + (lane & 15);
    int loff = (lane & 16) ? 8 : 0;
    size_t rb = (size_t)row * F_DIM;

    // build all 4 A k-step fragments once
    v16h af[4];
#pragma unroll
    for (int ks = 0; ks < 4; ++ks) {
        int kb = ks * 32 + loff;
        float4 A0 = *(const float4*)(p1 + rb + kb);
        float4 A1 = *(const float4*)(p1 + rb + kb + 4);
        float4 A2 = *(const float4*)(p1 + rb + kb + 16);
        float4 A3 = *(const float4*)(p1 + rb + kb + 20);
        if (p2) {
            float4 C0 = *(const float4*)(p2 + rb + kb);
            float4 C1 = *(const float4*)(p2 + rb + kb + 4);
            float4 C2 = *(const float4*)(p2 + rb + kb + 16);
            float4 C3 = *(const float4*)(p2 + rb + kb + 20);
            A0.x -= C0.x; A0.y -= C0.y; A0.z -= C0.z; A0.w -= C0.w;
            A1.x -= C1.x; A1.y -= C1.y; A1.z -= C1.z; A1.w -= C1.w;
            A2.x -= C2.x; A2.y -= C2.y; A2.z -= C2.z; A2.w -= C2.w;
            A3.x -= C3.x; A3.y -= C3.y; A3.z -= C3.z; A3.w -= C3.w;
        }
        v16h a;
        a[0]  = (_Float16)A0.x; a[1]  = (_Float16)A0.y; a[2]  = (_Float16)A0.z; a[3]  = (_Float16)A0.w;
        a[4]  = (_Float16)A1.x; a[5]  = (_Float16)A1.y; a[6]  = (_Float16)A1.z; a[7]  = (_Float16)A1.w;
        a[8]  = (_Float16)A2.x; a[9]  = (_Float16)A2.y; a[10] = (_Float16)A2.z; a[11] = (_Float16)A2.w;
        a[12] = (_Float16)A3.x; a[13] = (_Float16)A3.y; a[14] = (_Float16)A3.z; a[15] = (_Float16)A3.w;
        af[ks] = a;
    }

    const _Float16* wbase = wp + (size_t)c * 8 * 4 * 32 * 16;
    int mb   = m0 + ((lane & 16) ? 8 : 0);
    int colr = lane & 15;

#pragma unroll
    for (int nt = 0; nt < 8; ++nt) {
        v8f acc = {0.f, 0.f, 0.f, 0.f, 0.f, 0.f, 0.f, 0.f};
#pragma unroll
        for (int ks = 0; ks < 4; ++ks) {
            v16h bf = *(const v16h*)(wbase + (((size_t)nt * 4 + ks) * 32 + lane) * 16);
            acc = __builtin_amdgcn_wmma_f32_16x16x32_f16(false, af[ks], false, bf,
                                                         (short)0, acc, false, false);
        }
        int col = nt * 16 + colr;
        float bias = (c < 4) ? blow[c * O_DIM + col] : bband[(c - 4) * O_DIM + col];
#pragma unroll
        for (int r = 0; r < 8; ++r)
            hH[((size_t)c * N + (mb + r)) * O_DIM + col] = (_Float16)(acc[r] + bias);
    }
}

// ---------------- channel attention ----------------
__device__ __forceinline__ float blk_reduce(float v, volatile float* red) {
    int t = threadIdx.x;
    red[t] = v;
    __syncthreads();
#pragma unroll
    for (int s = 64; s > 0; s >>= 1) {
        if (t < s) red[t] = red[t] + red[t + s];
        __syncthreads();
    }
    float r = red[0];
    __syncthreads();
    return r;
}

__global__ void k_att(const _Float16* __restrict__ hH,
                      const float* __restrict__ apl, const float* __restrict__ acl,
                      const float* __restrict__ apb, const float* __restrict__ acb,
                      _Float16* __restrict__ cc, int N) {
    __shared__ float red[128];
    int n = blockIdx.x, o = threadIdx.x;
    size_t NO = (size_t)N * O_DIM;
    size_t base = (size_t)n * O_DIM + o;

    float hl[4], al[4], hb[3], ab[3];
#pragma unroll
    for (int c = 0; c < 4; ++c) { hl[c] = (float)hH[c * NO + base]; al[c] = fmaxf(hl[c], 0.f); }
#pragma unroll
    for (int c = 0; c < 3; ++c) { hb[c] = (float)hH[(4 + c) * NO + base]; ab[c] = fmaxf(hb[c], 0.f); }

    float hpre_l = (al[0] + al[1] + al[2] + al[3]) * 0.25f;
    float hpre_b = (ab[0] + ab[1] + ab[2]) * (1.f / 3.f);
    float vapl = apl[o], vacl = acl[o], vapb = apb[o], vacb = acb[o];

    float dpre_l = blk_reduce(hpre_l * vapl, red);
    float dcl[4];
#pragma unroll
    for (int c = 0; c < 4; ++c) dcl[c] = blk_reduce(al[c] * vacl, red);
    float dpre_b = blk_reduce(hpre_b * vapb, red);
    float dcb[3];
#pragma unroll
    for (int c = 0; c < 3; ++c) dcb[c] = blk_reduce(ab[c] * vacb, red);

    // softmax over low channels
    float lg[4];
#pragma unroll
    for (int c = 0; c < 4; ++c) lg[c] = fmaxf(dpre_l + dcl[c], 0.f);
    float mx = fmaxf(fmaxf(lg[0], lg[1]), fmaxf(lg[2], lg[3]));
    float ex[4], s = 0.f;
#pragma unroll
    for (int c = 0; c < 4; ++c) { ex[c] = __expf(lg[c] - mx); s += ex[c]; }
    float outl = 0.f;
#pragma unroll
    for (int c = 0; c < 4; ++c) outl += (ex[c] / s) * hl[c];

    // softmax over band channels
    float bg[3];
#pragma unroll
    for (int c = 0; c < 3; ++c) bg[c] = fmaxf(dpre_b + dcb[c], 0.f);
    float mxb = fmaxf(fmaxf(bg[0], bg[1]), bg[2]);
    float eb[3], sb = 0.f;
#pragma unroll
    for (int c = 0; c < 3; ++c) { eb[c] = __expf(bg[c] - mxb); sb += eb[c]; }
    float outb = 0.f;
#pragma unroll
    for (int c = 0; c < 3; ++c) outb += (eb[c] / sb) * hb[c];

    cc[(size_t)n * 256 + o]       = (_Float16)outl;
    cc[(size_t)n * 256 + 128 + o] = (_Float16)outb;
}

// ---------------- final MLP GEMM: [N,256] f16 @ [256,128] -> f32 out ----------------
// A fragments (8 ksteps) held in registers, reused across all 8 N tiles.
__global__ void k_gemm_mlp(const _Float16* __restrict__ A, const _Float16* __restrict__ wp,
                           const float* __restrict__ bias, float* __restrict__ out, int N) {
    int wave = threadIdx.x >> 5;
    int lane = threadIdx.x & 31;
    int m0 = (blockIdx.x * 4 + wave) * 16;
    if (m0 >= N) return;   // wave-uniform

    int row  = m0 + (lane & 15);
    int loff = (lane & 16) ? 8 : 0;
    const _Float16* ar = A + (size_t)row * 256;

    v16h af[8];
#pragma unroll
    for (int ks = 0; ks < 8; ++ks) {
        int kb = ks * 32 + loff;
        v8h lo = *(const v8h*)(ar + kb);
        v8h hi = *(const v8h*)(ar + kb + 16);
        af[ks] = __builtin_shufflevector(lo, hi, 0, 1, 2, 3, 4, 5, 6, 7,
                                         8, 9, 10, 11, 12, 13, 14, 15);
    }

    int mb   = m0 + ((lane & 16) ? 8 : 0);
    int colr = lane & 15;

#pragma unroll
    for (int nt = 0; nt < 8; ++nt) {
        v8f acc = {0.f, 0.f, 0.f, 0.f, 0.f, 0.f, 0.f, 0.f};
#pragma unroll
        for (int ks = 0; ks < 8; ++ks) {
            v16h bf = *(const v16h*)(wp + (((size_t)nt * 8 + ks) * 32 + lane) * 16);
            acc = __builtin_amdgcn_wmma_f32_16x16x32_f16(false, af[ks], false, bf,
                                                         (short)0, acc, false, false);
        }
        int col = nt * 16 + colr;
        float b = bias[col];
#pragma unroll
        for (int r = 0; r < 8; ++r)
            out[(size_t)(mb + r) * O_DIM + col] = acc[r] + b;
    }
}

// ---------------- host launcher ----------------
extern "C" void kernel_launch(void* const* d_in, const int* in_sizes, int n_in,
                              void* d_out, int out_size, void* d_ws, size_t ws_size,
                              hipStream_t stream) {
    const int F = F_DIM, O = O_DIM;
    const int N = in_sizes[0] / F;
    const int E = in_sizes[1] / 2;

    const float*     x     = (const float*)d_in[0];
    const long long* ei    = (const long long*)d_in[1];
    const long long* src   = ei;
    const long long* dst   = ei + E;
    const float*     Wlow  = (const float*)d_in[2];
    const float*     blow  = (const float*)d_in[3];
    const float*     Wband = (const float*)d_in[4];
    const float*     bband = (const float*)d_in[5];
    const float*     apl   = (const float*)d_in[6];
    const float*     acl   = (const float*)d_in[7];
    const float*     apb   = (const float*)d_in[8];
    const float*     acb   = (const float*)d_in[9];
    const float*     Wmlp  = (const float*)d_in[10];
    const float*     bmlp  = (const float*)d_in[11];
    float*           out   = (float*)d_out;

    char* ws = (char*)d_ws;
    size_t off = 0;
    auto alloc = [&](size_t bytes) -> char* {
        char* p = ws + off;
        off = (off + bytes + 255) & ~((size_t)255);
        return p;
    };
    float* deg  = (float*)alloc((size_t)N * 4);
    float* dinv = (float*)alloc((size_t)N * 4);
    float* wE   = (float*)alloc((size_t)E * 4);
    float* agg[4];
    for (int i = 0; i < 4; ++i) agg[i] = (float*)alloc((size_t)N * F * 4);
    _Float16* hH  = (_Float16*)alloc((size_t)7 * N * O * 2);
    _Float16* cc  = (_Float16*)alloc((size_t)N * 2 * O * 2);
    _Float16* wpF = (_Float16*)alloc((size_t)7 * F * O * 2);
    _Float16* wpM = (_Float16*)alloc((size_t)2 * O * O * 2);

    // edge weights
    k_zero<<<(N + 255) / 256, 256, 0, stream>>>(deg, N);
    k_deg<<<(E + 255) / 256, 256, 0, stream>>>(dst, deg, E);
    k_dinv<<<(N + 255) / 256, 256, 0, stream>>>(deg, dinv, N);
    k_edgew<<<(E + 255) / 256, 256, 0, stream>>>(src, dst, dinv, wE, E);

    // 4 propagation hops
    const float* cur = x;
    for (int h = 0; h < 4; ++h) {
        float* nxt = agg[h];
        int n4 = N * F / 4;
        k_half<<<(n4 + 255) / 256, 256, 0, stream>>>(cur, nxt, n4);
        k_scatter<<<(E + 7) / 8, 256, 0, stream>>>(src, dst, wE, cur, nxt, E);
        cur = nxt;
    }

    // pack weights into fragment layout
    k_packW<<<(7 * F * O + 255) / 256, 256, 0, stream>>>(Wlow, Wband, wpF, 7 * F * O);
    k_packWmlp<<<(2 * O * O + 255) / 256, 256, 0, stream>>>(Wmlp, wpM, 2 * O * O);

    // per-channel GEMMs (one wave per 16-row M tile, all 8 N tiles)
    int mtiles = (N + 15) / 16;
    dim3 g1((mtiles + 3) / 4, 7, 1);
    k_gemm_feats<<<g1, 128, 0, stream>>>(x, agg[0], agg[1], agg[3], wpF, blow, bband, hH, N);

    // attention combine
    k_att<<<N, 128, 0, stream>>>(hH, apl, acl, apb, acb, cc, N);

    // final MLP
    dim3 g2((mtiles + 3) / 4, 1, 1);
    k_gemm_mlp<<<g2, 128, 0, stream>>>(cc, wpM, bmlp, out, N);
}
